// LatticeMultiHeadAttention_25400436589158
// MI455X (gfx1250) — matrix operand order repaired and checked by reading, not compiled, not run on hardware
//
#include <hip/hip_runtime.h>
#include <hip/hip_bf16.h>
#include <math.h>

// ---------------------------------------------------------------------------
// Problem constants (B=4, S=1024, D=1024, H=16, DK=64)
// ---------------------------------------------------------------------------
#define BB 4
#define SS 1024
#define DD 1024
#define HH 16
#define DKK 64

typedef __attribute__((ext_vector_type(16))) __bf16 v16bf;
typedef __attribute__((ext_vector_type(8)))  float  v8f;
typedef __attribute__((ext_vector_type(4)))  unsigned int v4u;

struct u32x8 { v4u a, b; };

__device__ __forceinline__ unsigned short f2bf(float f) {
    unsigned int u = __float_as_uint(f);
    u += 0x7FFFu + ((u >> 16) & 1u);          // round-to-nearest-even
    return (unsigned short)(u >> 16);
}

// 16 bf16 elements from two 16-byte-aligned halves -> pure b128 loads, no VALU
__device__ __forceinline__ v16bf ld16bf(const unsigned short* p0,
                                        const unsigned short* p1) {
    u32x8 t;
    t.a = *(const v4u*)p0;
    t.b = *(const v4u*)p1;
    return __builtin_bit_cast(v16bf, t);
}

// ---------------------------------------------------------------------------
// Kernel 0: elementwise f32 -> bf16 cast (8 elements / thread, b128 in/out)
// ---------------------------------------------------------------------------
__global__ __launch_bounds__(256) void cast_bf16_kernel(const float* __restrict__ src,
                                                        unsigned int* __restrict__ dst) {
    const int i = blockIdx.x * 256 + threadIdx.x;
    const float4 f0 = ((const float4*)src)[2 * i];
    const float4 f1 = ((const float4*)src)[2 * i + 1];
    v4u o;
    o.x = ((unsigned)f2bf(f0.y) << 16) | f2bf(f0.x);
    o.y = ((unsigned)f2bf(f0.w) << 16) | f2bf(f0.z);
    o.z = ((unsigned)f2bf(f1.y) << 16) | f2bf(f1.x);
    o.w = ((unsigned)f2bf(f1.w) << 16) | f2bf(f1.z);
    ((v4u*)dst)[i] = o;
}

// ---------------------------------------------------------------------------
// Kernel 1: sequential Kuramoto phase update -> bias[h] = 0.1*cos(phase_h)
// ---------------------------------------------------------------------------
__global__ void phase_bias_kernel(const float* __restrict__ freq,
                                  const float* __restrict__ phase0,
                                  const float* __restrict__ coup,
                                  const float* __restrict__ latc,
                                  float* __restrict__ bias) {
    if (threadIdx.x == 0 && blockIdx.x == 0) {
        float ph[HH];
        for (int h = 0; h < HH; ++h) ph[h] = phase0[h];
        for (int h = 0; h < HH; ++h) {
            int px = h >> 2, py = h & 3;
            float force = 0.0f;
            for (int o = 0; o < HH; ++o) {
                if (o == h) continue;
                int qx = o >> 2, qy = o & 3;
                int d2 = (px - qx) * (px - qx) + (py - qy) * (py - qy);
                if (d2 <= 2) {                  // dist <= sqrt(2)+0.1
                    float dist = sqrtf((float)d2);
                    float w = expf(-dist / latc[h]);
                    force += w * sinf(ph[o] - ph[h]);
                }
            }
            ph[h] += 0.01f * (freq[h] + coup[h] * force);
        }
        for (int h = 0; h < HH; ++h) bias[h] = 0.1f * cosf(ph[h]);
    }
}

// ---------------------------------------------------------------------------
// Kernel 2: GEMM  C[M,N] = A[M,K] * W[N,K]^T, all-bf16 operands
//   (M=4096, N=K=1024).  Each wave: 32x64 tile = 8 WMMA / 32-K step.
//   MODE 0: store bf16 row-major [M,N]                (Q, K projections)
//   MODE 1: store bf16 transposed per batch [b][n][s] (V projection -> Vt)
//   MODE 2: store f32 [M,N] + bias[col]               (output projection)
// ---------------------------------------------------------------------------
template <int MODE>
__global__ __launch_bounds__(128) void gemm_nt_bf(const unsigned short* __restrict__ A,
                                                  const unsigned short* __restrict__ W,
                                                  void* __restrict__ outp,
                                                  const float* __restrict__ bias) {
    const int Kn = DD, Nn = DD;
    const int lane = threadIdx.x & 31;
    const int wv   = threadIdx.x >> 5;
    const int lx   = lane & 15;
    const int hi   = lane >> 4;
    const int m0   = blockIdx.x * 32;
    const int n0   = blockIdx.y * 256 + wv * 64;

    v8f acc[2][4] = {};
    size_t abase[2];
#pragma unroll
    for (int ar = 0; ar < 2; ++ar)
        abase[ar] = (size_t)(m0 + ar * 16 + lx) * Kn;

    for (int kb = 0; kb < Kn; kb += 32) {
        if (kb + 32 < Kn)
            __builtin_prefetch(&A[abase[0] + kb + 32], 0, 1);  // global_prefetch_b8

        v16bf a[2];
#pragma unroll
        for (int ar = 0; ar < 2; ++ar) {
            const size_t p = abase[ar] + kb + hi * 8;          // elems 0..7 / 8..15(+16)
            a[ar] = ld16bf(&A[p], &A[p + 16]);
        }
#pragma unroll
        for (int j = 0; j < 4; ++j) {
            const size_t wb = (size_t)(n0 + j * 16 + lx) * Kn + kb + hi * 8;
            const v16bf bf = ld16bf(&W[wb], &W[wb + 16]);      // elems 0..7 / 8..15(+16)
#pragma unroll
            for (int ar = 0; ar < 2; ++ar)
                acc[ar][j] = __builtin_amdgcn_wmma_f32_16x16x32_bf16(
                    false, a[ar], false, bf, (short)0, acc[ar][j], false, false);
        }
    }

#pragma unroll
    for (int ar = 0; ar < 2; ++ar) {
#pragma unroll
        for (int j = 0; j < 4; ++j) {
#pragma unroll
            for (int r = 0; r < 8; ++r) {
                const int row = m0 + ar * 16 + hi * 8 + r;     // f32 C layout
                const int col = n0 + j * 16 + lx;
                const float v = acc[ar][j][r];
                if (MODE == 0) {
                    ((unsigned short*)outp)[(size_t)row * Nn + col] = f2bf(v);
                } else if (MODE == 1) {
                    const int bb = row >> 10, s = row & 1023;  // row = b*S+s
                    ((unsigned short*)outp)[(((size_t)bb * Nn + col) << 10) + s] = f2bf(v);
                } else {
                    ((float*)outp)[(size_t)row * Nn + col] = v + bias[col];
                }
            }
        }
    }
}

// ---------------------------------------------------------------------------
// Kernel 3: attention per (b,h).  Each wave owns 16 query rows.
//  Pass 1: stream QK^T tiles (WMMA), online row max/sum.
//  Pass 2: recompute, write normalized p to attn (single 256MB pass),
//          transpose p via wave-private LDS into A-layout, PV via WMMA.
// ---------------------------------------------------------------------------
__global__ __launch_bounds__(128) void attn_kernel(const unsigned short* __restrict__ Qbf,
                                                   const unsigned short* __restrict__ Kbf,
                                                   const unsigned short* __restrict__ Vt,
                                                   const float* __restrict__ biasv,
                                                   float* __restrict__ attn_out,
                                                   unsigned short* __restrict__ Ctx) {
    __shared__ unsigned short lds_p[4][16][32];   // per-wave P staging tile

    const int lane = threadIdx.x & 31;
    const int wv   = threadIdx.x >> 5;
    const int lx   = lane & 15;
    const int hi   = lane >> 4;
    const int bh   = blockIdx.x;                  // b*H + h
    const int b    = bh >> 4;
    const int h    = bh & 15;
    const int s0   = blockIdx.y * 64 + wv * 16;
    const float scale = 0.125f;                   // 1/sqrt(64)
    const float bias  = biasv[h];

    // Q fragments for the wave's 16 rows over DK=64 (two 16x32 A-frags)
    v16bf qa[2];
    {
        const size_t qbase = ((size_t)(b * SS + s0 + lx)) * DD + h * DKK;
#pragma unroll
        for (int kk = 0; kk < 2; ++kk) {
            const size_t p = qbase + kk * 32 + hi * 8;
            qa[kk] = ld16bf(&Qbf[p], &Qbf[p + 16]);
        }
    }

    // ---- pass 1: online max/sum over all S keys --------------------------
    float m8[8], l8[8];
#pragma unroll
    for (int r = 0; r < 8; ++r) { m8[r] = -1e30f; l8[r] = 0.0f; }

    for (int t0 = 0; t0 < SS; t0 += 32) {
#pragma unroll
        for (int half = 0; half < 2; ++half) {
            const int kcol = t0 + half * 16 + lx;
            const size_t kbase = ((size_t)(b * SS + kcol)) * DD + h * DKK;
            v8f sc = {};
#pragma unroll
            for (int kk = 0; kk < 2; ++kk) {
                const size_t p = kbase + kk * 32 + hi * 8;
                const v16bf kf = ld16bf(&Kbf[p], &Kbf[p + 16]);
                sc = __builtin_amdgcn_wmma_f32_16x16x32_bf16(
                    false, qa[kk], false, kf, (short)0, sc, false, false);
            }
#pragma unroll
            for (int r = 0; r < 8; ++r) {
                const float v  = sc[r] * scale + bias;
                const float mn = fmaxf(m8[r], v);
                l8[r] = l8[r] * __expf(m8[r] - mn) + __expf(v - mn);
                m8[r] = mn;
            }
        }
    }
    // reduce (m,l) across the 16 lanes of each half-wave group
    float rowM[8], rowInv[8];
#pragma unroll
    for (int r = 0; r < 8; ++r) {
        float M = m8[r];
#pragma unroll
        for (int o = 1; o < 16; o <<= 1) M = fmaxf(M, __shfl_xor(M, o, 16));
        float sw = l8[r] * __expf(m8[r] - M);
#pragma unroll
        for (int o = 1; o < 16; o <<= 1) sw += __shfl_xor(sw, o, 16);
        rowM[r]   = M;
        rowInv[r] = 1.0f / sw;
    }

    // ---- pass 2: write normalized attn + accumulate ctx = P @ V ----------
    v8f ctx[4] = {};
    float* attn0 = attn_out + ((size_t)bh * SS) * SS;

    for (int t0 = 0; t0 < SS; t0 += 32) {
#pragma unroll
        for (int half = 0; half < 2; ++half) {
            const int kcol = t0 + half * 16 + lx;
            const size_t kbase = ((size_t)(b * SS + kcol)) * DD + h * DKK;
            v8f sc = {};
#pragma unroll
            for (int kk = 0; kk < 2; ++kk) {
                const size_t p = kbase + kk * 32 + hi * 8;
                const v16bf kf = ld16bf(&Kbf[p], &Kbf[p + 16]);
                sc = __builtin_amdgcn_wmma_f32_16x16x32_bf16(
                    false, qa[kk], false, kf, (short)0, sc, false, false);
            }
#pragma unroll
            for (int r = 0; r < 8; ++r) {
                const float v = sc[r] * scale + bias;
                const float p = __expf(v - rowM[r]) * rowInv[r];
                const int  row = s0 + hi * 8 + r;
                attn0[(size_t)row * SS + kcol] = p;                 // final attn
                lds_p[wv][hi * 8 + r][half * 16 + lx] = f2bf(p);    // C->A stage
            }
        }
        asm volatile("s_wait_dscnt 0" ::: "memory");   // intra-wave LDS RAW

        // P as A-fragment (16 rows x 32 keys, ISA 16-bit A layout)
        const v16bf pa = ld16bf(&lds_p[wv][lx][hi * 8],
                                &lds_p[wv][lx][16 + hi * 8]);
        // V as B-fragments (32 keys x 16 e), contiguous thanks to Vt layout
#pragma unroll
        for (int j = 0; j < 4; ++j) {
            const size_t vbase =
                ((size_t)b * DD + h * DKK + j * 16 + lx) * SS + t0 + hi * 8;
            const v16bf vf = ld16bf(&Vt[vbase], &Vt[vbase + 16]);
            ctx[j] = __builtin_amdgcn_wmma_f32_16x16x32_bf16(
                false, pa, false, vf, (short)0, ctx[j], false, false);
        }
    }

    // store ctx tile into head-concatenated buffer (bf16) for output GEMM
#pragma unroll
    for (int j = 0; j < 4; ++j) {
#pragma unroll
        for (int r = 0; r < 8; ++r) {
            const int row = s0 + hi * 8 + r;
            const int col = h * DKK + j * 16 + lx;
            Ctx[(size_t)row * DD + col] = f2bf(ctx[j][r]);
        }
    }
}

// ---------------------------------------------------------------------------
// Host-side launch
// d_ws (u16 elems): Qbf|Kbf|Vt|Ctx (4M each) | Xq|Xk|Xv (4M each)
//                   | Wqb|Wkb|Wvb|Wob (1M each) | bias f32[16]   (~64 MB)
// d_out: out f32[4M] then attn f32[64M]
// ---------------------------------------------------------------------------
extern "C" void kernel_launch(void* const* d_in, const int* in_sizes, int n_in,
                              void* d_out, int out_size, void* d_ws, size_t ws_size,
                              hipStream_t stream) {
    const float* query = (const float*)d_in[0];
    const float* key   = (const float*)d_in[1];
    const float* value = (const float*)d_in[2];
    const float* Wq    = (const float*)d_in[3];
    const float* Wk    = (const float*)d_in[4];
    const float* Wv    = (const float*)d_in[5];
    const float* freq  = (const float*)d_in[6];
    const float* phase = (const float*)d_in[7];
    const float* coup  = (const float*)d_in[8];
    const float* latc  = (const float*)d_in[9];
    const float* Wo    = (const float*)d_in[10];
    const float* bo    = (const float*)d_in[11];

    const size_t MN = (size_t)BB * SS * DD;       // 4M elements
    const size_t WN = (size_t)DD * DD;            // 1M elements
    unsigned short* ws16 = (unsigned short*)d_ws;
    unsigned short* Qbf = ws16;
    unsigned short* Kbf = Qbf + MN;
    unsigned short* Vt  = Kbf + MN;
    unsigned short* Ctx = Vt + MN;
    unsigned short* Xq  = Ctx + MN;
    unsigned short* Xk  = Xq + MN;
    unsigned short* Xv  = Xk + MN;
    unsigned short* Wqb = Xv + MN;
    unsigned short* Wkb = Wqb + WN;
    unsigned short* Wvb = Wkb + WN;
    unsigned short* Wob = Wvb + WN;
    float*          bias = (float*)(Wob + WN);

    float* out_f  = (float*)d_out;
    float* attn_f = out_f + MN;

    phase_bias_kernel<<<1, 32, 0, stream>>>(freq, phase, coup, latc, bias);

    // pre-cast activations + weights to bf16 (8 elems/thread)
    const int gx = (int)(MN / 8 / 256);           // 2048 blocks
    const int gw = (int)(WN / 8 / 256);           // 512 blocks
    cast_bf16_kernel<<<gx, 256, 0, stream>>>(query, (unsigned int*)Xq);
    cast_bf16_kernel<<<gx, 256, 0, stream>>>(key,   (unsigned int*)Xk);
    cast_bf16_kernel<<<gx, 256, 0, stream>>>(value, (unsigned int*)Xv);
    cast_bf16_kernel<<<gw, 256, 0, stream>>>(Wq,    (unsigned int*)Wqb);
    cast_bf16_kernel<<<gw, 256, 0, stream>>>(Wk,    (unsigned int*)Wkb);
    cast_bf16_kernel<<<gw, 256, 0, stream>>>(Wv,    (unsigned int*)Wvb);
    cast_bf16_kernel<<<gw, 256, 0, stream>>>(Wo,    (unsigned int*)Wob);

    dim3 gemm_grid(BB * SS / 32, DD / 256);       // (128, 4)
    dim3 attn_grid(BB * HH, SS / 64);             // (64, 16)

    gemm_nt_bf<0><<<gemm_grid, 128, 0, stream>>>(Xq, Wqb, (void*)Qbf, nullptr);
    gemm_nt_bf<0><<<gemm_grid, 128, 0, stream>>>(Xk, Wkb, (void*)Kbf, nullptr);
    gemm_nt_bf<1><<<gemm_grid, 128, 0, stream>>>(Xv, Wvb, (void*)Vt,  nullptr);

    attn_kernel<<<attn_grid, 128, 0, stream>>>(Qbf, Kbf, Vt, bias, attn_f, Ctx);

    gemm_nt_bf<2><<<gemm_grid, 128, 0, stream>>>(Ctx, Wob, (void*)out_f, bo);
}
